// RULPredictionModel_26843545600120
// MI455X (gfx1250) — matrix operand order repaired and checked by reading, not compiled, not run on hardware
//
#include <hip/hip_runtime.h>
#include <hip/hip_bf16.h>

// =====================================================================
// MI455X (gfx1250, wave32) implementation of the 3-layer MoE transformer
// regressor. All GEMM-shaped math uses v_wmma_f32_16x16x32_f16 (f16
// operands, fp32 accumulate). GEMM tiles are staged by the Tensor Data
// Mover (tensor_load_to_lds, TENSORcnt-pipelined); attention tiles use
// async global->LDS b128 copies (ASYNCcnt).
// =====================================================================

#define DEV __device__ __forceinline__

typedef __attribute__((ext_vector_type(16))) _Float16 v16h;
typedef __attribute__((ext_vector_type(8)))  float    v8f;
typedef __attribute__((ext_vector_type(4)))  int      v4i;
typedef unsigned int u32;

// model dims
constexpr int B_  = 16, S_ = 1024, D_ = 384, H_ = 8, L_ = 3, E_ = 8, FF_ = 768;
constexpr int DH_ = D_ / H_;       // 48
constexpr int T_  = B_ * S_;       // 16384 tokens
constexpr int D3_ = 3 * D_;        // 1152

// ---------------- CDNA5 async global->LDS copy (16B per lane) ----------
#if defined(__has_builtin)
#if __has_builtin(__builtin_amdgcn_global_load_async_to_lds_b128)
#define ASYNC_COPY_OK 1
#endif
#if __has_builtin(__builtin_amdgcn_s_wait_asynccnt)
#define ASYNC_WAIT_BUILTIN 1
#endif
#if __has_builtin(__builtin_amdgcn_tensor_load_to_lds) && \
    __has_builtin(__builtin_amdgcn_s_wait_tensorcnt)
#define TDM_OK 1
#endif
#endif
#ifndef ASYNC_COPY_OK
#define ASYNC_COPY_OK 0
#endif
#ifndef ASYNC_WAIT_BUILTIN
#define ASYNC_WAIT_BUILTIN 0
#endif
#ifndef TDM_OK
#define TDM_OK 0
#endif

DEV void async_cp16(const _Float16* g, _Float16* l) {
#if ASYNC_COPY_OK
  __builtin_amdgcn_global_load_async_to_lds_b128(
      (__attribute__((address_space(1))) v4i*)(g),
      (__attribute__((address_space(3))) v4i*)(l), 0, 0);
#else
  *(uint4*)l = *(const uint4*)g;
#endif
}
DEV void async_wait0() {
#if ASYNC_COPY_OK
#if ASYNC_WAIT_BUILTIN
  __builtin_amdgcn_s_wait_asynccnt(0);
#else
  asm volatile("s_wait_asynccnt 0x0" ::: "memory");
#endif
#endif
}
DEV void async_wait4() {
#if ASYNC_COPY_OK
#if ASYNC_WAIT_BUILTIN
  __builtin_amdgcn_s_wait_asynccnt(4);
#else
  asm volatile("s_wait_asynccnt 0x4" ::: "memory");
#endif
#endif
}

// ---------------- CDNA5 Tensor Data Mover: 2D tile, 2-byte elements ----
#if TDM_OK
typedef __attribute__((ext_vector_type(4))) unsigned int v4u;
typedef __attribute__((ext_vector_type(8))) int v8i_t;

// D# per CDNA5 ISA: group0 = {flags, lds_addr,
// global_addr, type=2}; group1 = {data_size/pad, tensor dims, tile dims,
// dim0 stride}. Groups 2/3 zero (2D tensor). Padding codes: interval
// 2^(c+1) DWORDs, amount (c+1) DWORDs. 6-arg builtin form (this
// toolchain): (g0, g1, g2, g3, g4, cpol), trailing groups zero-filled.
DEV void tdm_load_2d(const _Float16* gsrc, _Float16* ldst,
                     u32 tile_x, u32 tile_y, u32 row_stride,
                     u32 pad_interval_code, u32 pad_amount_code) {
  const unsigned long long ga = (unsigned long long)(size_t)gsrc;
  const u32 lds =
      (u32)(size_t)(__attribute__((address_space(3))) _Float16*)ldst;
  v4u g0;
  g0[0] = 1u;                                              // count=1, user mode
  g0[1] = lds;                                             // lds_addr
  g0[2] = (u32)ga;                                         // global_addr[31:0]
  g0[3] = (u32)((ga >> 32) & 0x01FFFFFFull) | (2u << 30);  // addr[56:32]|type=2
  v8i_t g1;
  g1[0] = (int)((1u << 16) |                               // data_size = 2B
                (1u << 20) |                               // pad_enable
                (pad_interval_code << 22) | (pad_amount_code << 25));
  g1[1] = (int)((row_stride & 0xFFFFu) << 16);             // tensor_dim0 lo16
  g1[2] = (int)(((row_stride >> 16) & 0xFFFFu) |           // tensor_dim0 hi16
                ((tile_y & 0xFFFFu) << 16));               // tensor_dim1 lo16
  g1[3] = (int)((tile_x & 0xFFFFu) << 16);                 // dim1 hi=0|tile_dim0
  g1[4] = (int)(tile_y & 0xFFFFu);                         // tile_dim1, dim2=0
  g1[5] = (int)row_stride;                                 // dim0_stride lo32
  g1[6] = 0;                                               // stride hi, d1s lo
  g1[7] = 0;
  const v4i z4 = {0, 0, 0, 0};
  const v8i_t z8 = {0, 0, 0, 0, 0, 0, 0, 0};
  __builtin_amdgcn_tensor_load_to_lds(g0, g1, z4, z4, z8, 0);
}
#endif

DEV v8f wmma32(v16h a, v16h b, v8f c) {
  return __builtin_amdgcn_wmma_f32_16x16x32_f16(
      /*neg_a=*/false, a, /*neg_b=*/false, b,
      /*c_mod=*/(short)0, c, /*reuse_a=*/false, /*reuse_b=*/false);
}

// A fragment (16x32 f16), ISA layout: lane L -> M = L&15,
// K = (j>>3)*16 + (L>>4)*8 + (j&7)  for half index j in [0,16)
DEV v16h frag_a(const _Float16* s, int ld, int m0, int k0) {
  const int lane = threadIdx.x & 31;
  const _Float16* p = s + (m0 + (lane & 15)) * ld + k0 + ((lane >> 4) << 3);
  v16h a;
#pragma unroll
  for (int j = 0; j < 16; ++j) a[j] = p[((j >> 3) << 4) + (j & 7)];
  return a;
}

// B fragment (32x16 f16), ISA layout: lane L -> N = L&15,
// K = (j>>3)*16 + (L>>4)*8 + (j&7)  for half index j in [0,16)
DEV v16h frag_b(const _Float16* s, int ld, int k0, int n0) {
  const int lane = threadIdx.x & 31;
  const _Float16* p = s + (k0 + ((lane >> 4) << 3)) * ld + n0 + (lane & 15);
  v16h b;
#pragma unroll
  for (int j = 0; j < 16; ++j) b[j] = p[(((j >> 3) << 4) + (j & 7)) * ld];
  return b;
}

#define ZERO8 {0.f,0.f,0.f,0.f,0.f,0.f,0.f,0.f}

// ---------------------------------------------------------------------
// Generic WMMA GEMM:  out = epilogue(A[M,K] @ B[K,N] + bias)
// Block tile 128x128, BK=32, 256 threads = 8 waves (4x2); 32x64 per wave
// (8 v_wmma per K-step). Double-buffered LDS staged by the TDM (wave 0
// issues 2 tensor_load_to_lds per stage, pipelined on TENSORcnt); async
// b128 copies as fallback. TDM descriptor padding reproduces the
// bank-conflict LDS strides (A: 16+4 DWORD rows, B: 64+4 DWORD rows).
// M % 128 == 0, N % 128 == 0, K % 32 == 0 (all shapes here comply).
// ---------------------------------------------------------------------
constexpr int LDA = 40;   // halfs; 80B rows -> 16B aligned
constexpr int LDB = 136;  // halfs; 272B rows -> 16B aligned

template <bool OUT_HALF, bool RELU, bool ACC, bool ROWSCALE>
__global__ __launch_bounds__(256) void gemm_kernel(
    const _Float16* __restrict__ A, const _Float16* __restrict__ Bw,
    const float* __restrict__ bias, const float* __restrict__ rowscale,
    int rs_stride, float* __restrict__ Cf, _Float16* __restrict__ Ch,
    int M, int N, int K) {
  (void)M;
  __shared__ __align__(16) _Float16 As[2][128 * LDA];
  __shared__ __align__(16) _Float16 Bs[2][32 * LDB];
  const int tid = threadIdx.x;
  const int wid = tid >> 5;
  const int wm  = wid >> 1;           // 0..3  (32 rows)
  const int wn  = wid & 1;            // 0..1  (64 cols)
  const int m0  = blockIdx.y * 128;
  const int n0  = blockIdx.x * 128;

  v8f c[2][4];
#pragma unroll
  for (int i = 0; i < 2; ++i)
#pragma unroll
    for (int j = 0; j < 4; ++j) c[i][j] = (v8f)ZERO8;

#if TDM_OK
  // wave 0 drives the Tensor Data Mover: one 128x32 A tile + one 32x128
  // B tile per stage (2 TENSORcnt increments).
  auto stage = [&](int kt, int bfi) {
    if (wid == 0) {
      tdm_load_2d(A + (size_t)m0 * K + kt, &As[bfi][0],
                  /*tile_x=*/32u, /*tile_y=*/128u, (u32)K, 3u, 3u);
      tdm_load_2d(Bw + (size_t)kt * N + n0, &Bs[bfi][0],
                  /*tile_x=*/128u, /*tile_y=*/32u, (u32)N, 5u, 3u);
    }
  };
#else
  // async-copy staging: 2 A-chunk + 2 B-chunk b128 copies per thread
  auto stage = [&](int kt, int bfi) {
#pragma unroll
    for (int i = 0; i < 2; ++i) {               // A: 128x32 halfs, 512 x 16B
      const int idx = tid + i * 256;
      const int r = idx >> 2, c8 = (idx & 3) << 3;
      async_cp16(A + (size_t)(m0 + r) * K + kt + c8, &As[bfi][r * LDA + c8]);
    }
#pragma unroll
    for (int i = 0; i < 2; ++i) {               // B: 32x128 halfs, 512 x 16B
      const int idx = tid + i * 256;
      const int r = idx >> 4, c8 = (idx & 15) << 3;
      async_cp16(Bw + (size_t)(kt + r) * N + n0 + c8, &Bs[bfi][r * LDB + c8]);
    }
  };
#endif

  stage(0, 0);
  int bf = 0;
  for (int kt = 0; kt < K; kt += 32, bf ^= 1) {
    const bool last = (kt + 32 >= K);
#if TDM_OK
    if (!last) {
      stage(kt + 32, bf ^ 1);
      if (wid == 0) __builtin_amdgcn_s_wait_tensorcnt(2);
    } else {
      if (wid == 0) __builtin_amdgcn_s_wait_tensorcnt(0);
    }
#else
    if (!last) { stage(kt + 32, bf ^ 1); async_wait4(); }
    else       { async_wait0(); }
#endif
    __syncthreads();
    const v16h a0 = frag_a(As[bf], LDA, wm * 32, 0);
    const v16h a1 = frag_a(As[bf], LDA, wm * 32 + 16, 0);
#pragma unroll
    for (int j = 0; j < 4; ++j) {
      const v16h b = frag_b(Bs[bf], LDB, 0, wn * 64 + j * 16);
      c[0][j] = wmma32(a0, b, c[0][j]);
      c[1][j] = wmma32(a1, b, c[1][j]);
    }
    __syncthreads();
  }

  const int lane  = tid & 31;
  const int rbase = m0 + wm * 32 + ((lane >> 4) << 3);
  const int cbase = n0 + wn * 64 + (lane & 15);
#pragma unroll
  for (int i = 0; i < 2; ++i)
#pragma unroll
    for (int j = 0; j < 4; ++j) {
#pragma unroll
      for (int r = 0; r < 8; ++r) {
        const int row = rbase + i * 16 + r;
        const int col = cbase + j * 16;
        float v = c[i][j][r] + bias[col];
        if (RELU) v = v > 0.f ? v : 0.f;
        if (ROWSCALE) v *= rowscale[(size_t)row * rs_stride];
        const size_t o = (size_t)row * N + col;
        if (OUT_HALF)      Ch[o] = (_Float16)v;
        else if (ACC)      Cf[o] += v;
        else               Cf[o] = v;
      }
    }
}

// ---------------------------------------------------------------------
// Flash attention, WMMA for QK^T and PV. One block = one (b,h) x 64-query
// tile; DH=48 padded to 64 with zeros. 256 threads = 8 waves.
// Q/V tiles staged via async b128 copies; K^T scattered (transpose).
// ---------------------------------------------------------------------
constexpr int LQ = 72;  // async-staged arrays: 144B rows (16B aligned)
constexpr int LV = 72;
constexpr int LK = 68;  // scalar-written arrays: conflict-free B/A frags
constexpr int LP = 68;

__global__ __launch_bounds__(256) void attn_kernel(
    const _Float16* __restrict__ qkv, _Float16* __restrict__ ao) {
  __shared__ __align__(16) _Float16 Qs[64 * LQ];
  __shared__ __align__(16) _Float16 Kts[64 * LK];  // transposed: [d][key]
  __shared__ __align__(16) _Float16 Vs[64 * LV];   // [key][d]
  __shared__ __align__(16) _Float16 Ps[64 * LP];   // probs, C->A relayout
  __shared__ float rowm[64], rowl[64], rowfac[64];
  __shared__ float pmax[2][64], psum[2][64];

  const int tid = threadIdx.x, lane = tid & 31, wid = tid >> 5;
  const int wq = wid >> 1;  // query row group (x16)
  const int wk = wid & 1;   // kv/d column group (x32)
  const int bh = blockIdx.y;
  const int b = bh / H_, hh = bh % H_;
  const int q0 = blockIdx.x * 64;
  const float scale = rsqrtf((float)DH_);

  const _Float16* qbase = qkv + ((size_t)b * S_) * D3_ + hh * DH_;

  // one-time zero padding: Qs/Vs cols 48..63, Kts rows 48..63
  for (int idx = tid; idx < 64 * 16; idx += 256) {
    const int r = idx >> 4, g = idx & 15;
    if (g >= 12) {
      _Float16* qd = Qs + r * LQ + g * 4;
      _Float16* vd = Vs + r * LV + g * 4;
      qd[0] = qd[1] = qd[2] = qd[3] = (_Float16)0.f;
      vd[0] = vd[1] = vd[2] = vd[3] = (_Float16)0.f;
    }
    if (r >= 48) Kts[r * LK + g * 4] = Kts[r * LK + g * 4 + 1] =
        Kts[r * LK + g * 4 + 2] = Kts[r * LK + g * 4 + 3] = (_Float16)0.f;
  }
  // stage Q tile (64 rows x 48 halfs = 384 x 16B chunks), async
  for (int idx = tid; idx < 64 * 6; idx += 256) {
    const int r = idx / 6, g8 = (idx % 6) << 3;
    async_cp16(qbase + (size_t)(q0 + r) * D3_ + g8, Qs + r * LQ + g8);
  }
  for (int i = tid; i < 64; i += 256) { rowm[i] = -3.0e38f; rowl[i] = 0.f; }

  v8f o0 = ZERO8, o1 = ZERO8;

  for (int kv0 = 0; kv0 < S_; kv0 += 64) {
    __syncthreads();  // protect Kts/Vs reuse across iterations
    // V tile: async 16B copies; K tile: load + transpose-scatter
    for (int idx = tid; idx < 64 * 6; idx += 256) {
      const int r = idx / 6, g8 = (idx % 6) << 3;
      async_cp16(qbase + 2 * D_ + (size_t)(kv0 + r) * D3_ + g8,
                 Vs + r * LV + g8);
    }
    for (int idx = tid; idx < 64 * 12; idx += 256) {
      const int rr = idx / 12, g = idx % 12;
      const uint2 kq =
          *(const uint2*)(qbase + D_ + (size_t)(kv0 + rr) * D3_ + g * 4);
      const _Float16* k4 = (const _Float16*)&kq;
#pragma unroll
      for (int i2 = 0; i2 < 4; ++i2) Kts[(g * 4 + i2) * LK + rr] = k4[i2];
    }
    async_wait0();
    __syncthreads();

    // S tile: 16q x 32kv per wave, K-dim = 64 (padded head dim)
    v8f s0 = ZERO8, s1 = ZERO8;
    {
      v16h a  = frag_a(Qs, LQ, wq * 16, 0);
      v16h b0 = frag_b(Kts, LK, 0, wk * 32);
      v16h b1 = frag_b(Kts, LK, 0, wk * 32 + 16);
      s0 = wmma32(a, b0, s0); s1 = wmma32(a, b1, s1);
      a  = frag_a(Qs, LQ, wq * 16, 32);
      b0 = frag_b(Kts, LK, 32, wk * 32);
      b1 = frag_b(Kts, LK, 32, wk * 32 + 16);
      s0 = wmma32(a, b0, s0); s1 = wmma32(a, b1, s1);
    }
    // scale + per-row partial max (reduce over 16 lanes holding one row)
    float pm[8];
#pragma unroll
    for (int r = 0; r < 8; ++r) {
      s0[r] *= scale; s1[r] *= scale;
      float m = fmaxf(s0[r], s1[r]);
#pragma unroll
      for (int o = 8; o; o >>= 1) m = fmaxf(m, __shfl_xor(m, o));
      pm[r] = m;
    }
    if ((lane & 15) == 0) {
      const int hf = lane >> 4;
#pragma unroll
      for (int r = 0; r < 8; ++r) pmax[wk][wq * 16 + hf * 8 + r] = pm[r];
    }
    __syncthreads();
    if (wk == 0 && lane < 16) {
      const int row = wq * 16 + lane;
      const float mo = rowm[row];
      const float mn = fmaxf(mo, fmaxf(pmax[0][row], pmax[1][row]));
      rowm[row] = mn;
      rowfac[row] = __expf(mo - mn);
    }
    __syncthreads();
    // p = exp(s - m), partial row sums, stage P (C-layout -> LDS -> A-layout)
    float ps[8];
#pragma unroll
    for (int r = 0; r < 8; ++r) {
      const int row = wq * 16 + ((lane >> 4) << 3) + r;
      const float mn = rowm[row];
      const float p0 = __expf(s0[r] - mn);
      const float p1 = __expf(s1[r] - mn);
      Ps[row * LP + wk * 32 + (lane & 15)]      = (_Float16)p0;
      Ps[row * LP + wk * 32 + 16 + (lane & 15)] = (_Float16)p1;
      float sm = p0 + p1;
#pragma unroll
      for (int o = 8; o; o >>= 1) sm += __shfl_xor(sm, o);
      ps[r] = sm;
    }
    if ((lane & 15) == 0) {
      const int hf = lane >> 4;
#pragma unroll
      for (int r = 0; r < 8; ++r) psum[wk][wq * 16 + hf * 8 + r] = ps[r];
    }
    __syncthreads();
    if (wk == 0 && lane < 16) {
      const int row = wq * 16 + lane;
      rowl[row] = rowl[row] * rowfac[row] + psum[0][row] + psum[1][row];
    }
    // O update: rescale then accumulate P @ V (K-dim = 64 keys)
#pragma unroll
    for (int r = 0; r < 8; ++r) {
      const float f = rowfac[wq * 16 + ((lane >> 4) << 3) + r];
      o0[r] *= f; o1[r] *= f;
    }
    {
      v16h a  = frag_a(Ps, LP, wq * 16, 0);
      v16h b0 = frag_b(Vs, LV, 0, wk * 32);
      v16h b1 = frag_b(Vs, LV, 0, wk * 32 + 16);
      o0 = wmma32(a, b0, o0); o1 = wmma32(a, b1, o1);
      a  = frag_a(Ps, LP, wq * 16, 32);
      b0 = frag_b(Vs, LV, 32, wk * 32);
      b1 = frag_b(Vs, LV, 32, wk * 32 + 16);
      o0 = wmma32(a, b0, o0); o1 = wmma32(a, b1, o1);
    }
  }
  __syncthreads();
  // normalize + write (head-concatenated [B,S,D] layout), drop padded cols
#pragma unroll
  for (int r = 0; r < 8; ++r) {
    const int row = wq * 16 + ((lane >> 4) << 3) + r;
    const float inv = 1.f / rowl[row];
    const int n_lo = wk * 32 + (lane & 15);
    const int n_hi = n_lo + 16;
    const size_t orow = ((size_t)b * S_ + q0 + row) * D_ + hh * DH_;
    if (n_lo < DH_) ao[orow + n_lo] = (_Float16)(o0[r] * inv);
    if (n_hi < DH_) ao[orow + n_hi] = (_Float16)(o1[r] * inv);
  }
}

// ---------------------------------------------------------------------
// LayerNorm (biased variance, matches jnp.var) : fp32 in -> f16 out
// ---------------------------------------------------------------------
__global__ __launch_bounds__(128) void ln_kernel(
    const float* __restrict__ x, const float* __restrict__ sc,
    const float* __restrict__ bi, _Float16* __restrict__ out) {
  const int t = blockIdx.x;
  const float* xr = x + (size_t)t * D_;
  const int i = threadIdx.x;
  const float v0 = xr[i], v1 = xr[i + 128], v2 = xr[i + 256];
  float s = v0 + v1 + v2, s2 = v0 * v0 + v1 * v1 + v2 * v2;
#pragma unroll
  for (int o = 16; o; o >>= 1) { s += __shfl_xor(s, o); s2 += __shfl_xor(s2, o); }
  __shared__ float sh[8];
  const int wid = i >> 5, lane = i & 31;
  if (lane == 0) { sh[wid] = s; sh[4 + wid] = s2; }
  __syncthreads();
  s  = sh[0] + sh[1] + sh[2] + sh[3];
  s2 = sh[4] + sh[5] + sh[6] + sh[7];
  const float mean = s / D_;
  const float var  = s2 / D_ - mean * mean;
  const float rs   = rsqrtf(var + 1e-5f);
  _Float16* orow = out + (size_t)t * D_;
  orow[i]       = (_Float16)((v0 - mean) * rs * sc[i]       + bi[i]);
  orow[i + 128] = (_Float16)((v1 - mean) * rs * sc[i + 128] + bi[i + 128]);
  orow[i + 256] = (_Float16)((v2 - mean) * rs * sc[i + 256] + bi[i + 256]);
}

// ---------------------------------------------------------------------
// Gating: softmax over E=8, top-2 renormalized gates, aux-loss sums.
// One wave per token.
// ---------------------------------------------------------------------
__global__ __launch_bounds__(256) void gate_kernel(
    const _Float16* __restrict__ xn, const float* __restrict__ gw,
    float* __restrict__ full, float* __restrict__ sums) {
  const int wid = threadIdx.x >> 5, lane = threadIdx.x & 31;
  const int t = blockIdx.x * 8 + wid;
  const _Float16* xr = xn + (size_t)t * D_;
  float acc[E_];
#pragma unroll
  for (int e = 0; e < E_; ++e) acc[e] = 0.f;
  for (int d = lane; d < D_; d += 32) {
    const float xv = (float)xr[d];
    const float* g = gw + (size_t)d * E_;
#pragma unroll
    for (int e = 0; e < E_; ++e) acc[e] += xv * g[e];
  }
#pragma unroll
  for (int e = 0; e < E_; ++e)
#pragma unroll
    for (int o = 16; o; o >>= 1) acc[e] += __shfl_xor(acc[e], o);
  if (lane == 0) {
    float mx = acc[0];
#pragma unroll
    for (int e = 1; e < E_; ++e) mx = fmaxf(mx, acc[e]);
    float p[E_], z = 0.f;
#pragma unroll
    for (int e = 0; e < E_; ++e) { p[e] = __expf(acc[e] - mx); z += p[e]; }
#pragma unroll
    for (int e = 0; e < E_; ++e) p[e] /= z;
    int i0 = 0;
#pragma unroll
    for (int e = 1; e < E_; ++e) if (p[e] > p[i0]) i0 = e;
    int i1 = (i0 == 0) ? 1 : 0;
#pragma unroll
    for (int e = 0; e < E_; ++e) if (e != i0 && p[e] > p[i1]) i1 = e;
    const float gs = p[i0] + p[i1];
    float* fr = full + (size_t)t * E_;
#pragma unroll
    for (int e = 0; e < E_; ++e) fr[e] = 0.f;
    fr[i0] = p[i0] / gs;
    fr[i1] = p[i1] / gs;
#pragma unroll
    for (int e = 0; e < E_; ++e) atomicAdd(&sums[e], p[e]);  // importance
    atomicAdd(&sums[8 + i0], 1.f);                            // load
    atomicAdd(&sums[8 + i1], 1.f);
  }
}

// ---------------------------------------------------------------------
// Attention-pooling regression head: one block per batch element.
// ---------------------------------------------------------------------
__global__ __launch_bounds__(256) void pool_kernel(
    const float* __restrict__ h, const float* __restrict__ att_w,
    const float* __restrict__ att_b, const float* __restrict__ head_w,
    const float* __restrict__ head_b, float* __restrict__ out) {
  __shared__ float logits[S_];
  __shared__ float red[256];
  const int b = blockIdx.x, tid = threadIdx.x;
  const float* hb = h + (size_t)b * S_ * D_;
  for (int s = tid; s < S_; s += 256) {
    const float* hr = hb + (size_t)s * D_;
    float acc = 0.f;
    for (int d = 0; d < D_; ++d) acc += hr[d] * att_w[d];
    logits[s] = acc + att_b[0];
  }
  __syncthreads();
  float m = -3.0e38f;
  for (int s = tid; s < S_; s += 256) m = fmaxf(m, logits[s]);
  red[tid] = m; __syncthreads();
  for (int st = 128; st; st >>= 1) {
    if (tid < st) red[tid] = fmaxf(red[tid], red[tid + st]);
    __syncthreads();
  }
  m = red[0]; __syncthreads();
  float z = 0.f;
  for (int s = tid; s < S_; s += 256) {
    const float e = __expf(logits[s] - m);
    logits[s] = e; z += e;
  }
  red[tid] = z; __syncthreads();
  for (int st = 128; st; st >>= 1) {
    if (tid < st) red[tid] += red[tid + st];
    __syncthreads();
  }
  z = red[0]; __syncthreads();
  const float rinv = 1.f / z;
  float part = 0.f;
  for (int d = tid; d < D_; d += 256) {
    float p = 0.f;
    for (int s = 0; s < S_; ++s) p += logits[s] * hb[(size_t)s * D_ + d];
    part += p * rinv * head_w[d];
  }
  red[tid] = part; __syncthreads();
  for (int st = 128; st; st >>= 1) {
    if (tid < st) red[tid] += red[tid + st];
    __syncthreads();
  }
  if (tid == 0) out[b] = red[0] + head_b[0];
}

__global__ void finalize_kernel(const float* __restrict__ sums,
                                float* __restrict__ out) {
  if (threadIdx.x == 0 && blockIdx.x == 0) {
    float tot = 0.f;
    for (int l = 0; l < L_; ++l) {
      float acc = 0.f;
      for (int e = 0; e < E_; ++e)
        acc += (sums[l * 16 + e] / (float)T_) * (sums[l * 16 + 8 + e] / (float)T_);
      tot += (float)E_ * acc;
    }
    out[16] = tot;
  }
}

__global__ void f32_to_f16_kernel(const float* __restrict__ in,
                                  _Float16* __restrict__ out, int n) {
  const int i = blockIdx.x * 256 + threadIdx.x;
  if (i < n) out[i] = (_Float16)in[i];
}

__global__ void proj_kernel(const float* __restrict__ x,
                            const float* __restrict__ pw,
                            const float* __restrict__ pb,
                            float* __restrict__ h) {
  const int i = blockIdx.x * 256 + threadIdx.x;  // over T_*D_
  const int t = i / D_, d = i % D_;
  h[i] = x[t] * pw[d] + pb[d];
}

// =====================================================================
extern "C" void kernel_launch(void* const* d_in, const int* in_sizes, int n_in,
                              void* d_out, int out_size, void* d_ws,
                              size_t ws_size, hipStream_t stream) {
  (void)in_sizes; (void)n_in; (void)out_size; (void)ws_size;
  const float* x      = (const float*)d_in[0];
  const float* proj_w = (const float*)d_in[1];
  const float* proj_b = (const float*)d_in[2];
  const float* qkv_w  = (const float*)d_in[3];
  const float* qkv_b  = (const float*)d_in[4];
  const float* out_w  = (const float*)d_in[5];
  const float* out_b  = (const float*)d_in[6];
  const float* n1_s   = (const float*)d_in[7];
  const float* n1_b   = (const float*)d_in[8];
  const float* n2_s   = (const float*)d_in[9];
  const float* n2_b   = (const float*)d_in[10];
  const float* gate_w = (const float*)d_in[11];
  const float* e_w1   = (const float*)d_in[12];
  const float* e_b1   = (const float*)d_in[13];
  const float* e_w2   = (const float*)d_in[14];
  const float* e_b2   = (const float*)d_in[15];
  const float* att_w  = (const float*)d_in[16];
  const float* att_b  = (const float*)d_in[17];
  const float* head_w = (const float*)d_in[18];
  const float* head_b = (const float*)d_in[19];
  float* out = (float*)d_out;

  // ---- workspace layout (~146 MB, 256B aligned regions) ----
  char* ws = (char*)d_ws;
  size_t off = 0;
  auto take = [&](size_t bytes) -> char* {
    char* p = ws + off;
    off = (off + bytes + 255) & ~(size_t)255;
    return p;
  };
  _Float16* qkvw16 = (_Float16*)take((size_t)L_ * D_ * D3_ * 2);
  _Float16* outw16 = (_Float16*)take((size_t)L_ * D_ * D_ * 2);
  _Float16* ew1_16 = (_Float16*)take((size_t)L_ * E_ * D_ * FF_ * 2);
  _Float16* ew2_16 = (_Float16*)take((size_t)L_ * E_ * FF_ * D_ * 2);
  float*    h      = (float*)take((size_t)T_ * D_ * 4);
  _Float16* xn16   = (_Float16*)take((size_t)T_ * D_ * 2);    // xn / xn2
  _Float16* qkv16  = (_Float16*)take((size_t)T_ * D3_ * 2);
  _Float16* ao16   = (_Float16*)take((size_t)T_ * D_ * 2);
  _Float16* hid16  = (_Float16*)take((size_t)T_ * FF_ * 2);
  float*    full   = (float*)take((size_t)T_ * E_ * 4);
  float*    sums   = (float*)take(L_ * 16 * 4);

  (void)hipMemsetAsync(sums, 0, L_ * 16 * sizeof(float), stream);

  // one-time-per-call f16 copies of the big weight tensors
  {
    const int n1 = L_ * D_ * D3_;
    f32_to_f16_kernel<<<(n1 + 255) / 256, 256, 0, stream>>>(qkv_w, qkvw16, n1);
    const int n2 = L_ * D_ * D_;
    f32_to_f16_kernel<<<(n2 + 255) / 256, 256, 0, stream>>>(out_w, outw16, n2);
    const int n3 = L_ * E_ * D_ * FF_;
    f32_to_f16_kernel<<<(n3 + 255) / 256, 256, 0, stream>>>(e_w1, ew1_16, n3);
    f32_to_f16_kernel<<<(n3 + 255) / 256, 256, 0, stream>>>(e_w2, ew2_16, n3);
  }

  // feature projection -> residual stream h (fp32)
  proj_kernel<<<(T_ * D_) / 256, 256, 0, stream>>>(x, proj_w, proj_b, h);

  for (int l = 0; l < L_; ++l) {
    // pre-norm attention
    ln_kernel<<<T_, 128, 0, stream>>>(h, n1_s + l * D_, n1_b + l * D_, xn16);
    gemm_kernel<true, false, false, false>
        <<<dim3(D3_ / 128, T_ / 128), 256, 0, stream>>>(
            xn16, qkvw16 + (size_t)l * D_ * D3_, qkv_b + l * D3_,
            nullptr, 0, nullptr, qkv16, T_, D3_, D_);
    attn_kernel<<<dim3(S_ / 64, B_ * H_), 256, 0, stream>>>(qkv16, ao16);
    gemm_kernel<false, false, true, false>
        <<<dim3(D_ / 128, T_ / 128), 256, 0, stream>>>(
            ao16, outw16 + (size_t)l * D_ * D_, out_b + l * D_,
            nullptr, 0, h, nullptr, T_, D_, D_);

    // pre-norm MoE FFN
    ln_kernel<<<T_, 128, 0, stream>>>(h, n2_s + l * D_, n2_b + l * D_, xn16);
    gate_kernel<<<T_ / 8, 256, 0, stream>>>(xn16, gate_w + (size_t)l * D_ * E_,
                                            full, sums + l * 16);
    for (int e = 0; e < E_; ++e) {
      gemm_kernel<true, true, false, false>
          <<<dim3(FF_ / 128, T_ / 128), 256, 0, stream>>>(
              xn16, ew1_16 + (size_t)(l * E_ + e) * D_ * FF_,
              e_b1 + (size_t)(l * E_ + e) * FF_, nullptr, 0, nullptr, hid16,
              T_, FF_, D_);
      gemm_kernel<false, false, true, true>
          <<<dim3(D_ / 128, T_ / 128), 256, 0, stream>>>(
              hid16, ew2_16 + (size_t)(l * E_ + e) * FF_ * D_,
              e_b2 + (size_t)(l * E_ + e) * D_, full + e, E_, h, nullptr,
              T_, D_, FF_);
    }
  }

  pool_kernel<<<B_, 256, 0, stream>>>(h, att_w, att_b, head_w, head_b, out);
  finalize_kernel<<<1, 32, 0, stream>>>(sums, out);
}
